// MRNNRL_simple_v5_80590766342943
// MI455X (gfx1250) — matrix operand order repaired and checked by reading, not compiled, not run on hardware
//
#include <hip/hip_runtime.h>
#include <hip/hip_bf16.h>

// Problem constants (from reference)
#define NN 300000
#define RR 1200
#define KK 20
#define BB 256
#define MM 50
#define DD 100
// derived
#define ROWS_PER_B 1000            // M*K
#define TILES_PER_B 63             // ceil(1000/16)
#define OUT_ACT 0
#define OUT_Q   256000
#define OUT_DOT 512000
#define OUT_NG  768000

typedef __attribute__((ext_vector_type(16))) __bf16 v16bf;
typedef __attribute__((ext_vector_type(8)))  __bf16 v8bf;
typedef __attribute__((ext_vector_type(8)))  float  v8f;
typedef __attribute__((ext_vector_type(4)))  unsigned int u32x4;
typedef __attribute__((ext_vector_type(4)))  int i32x4;
typedef __attribute__((ext_vector_type(8)))  int i32x8;

#if defined(__gfx1250__) && __has_builtin(__builtin_amdgcn_tensor_load_to_lds)
#define HAVE_TDM 1
#else
#define HAVE_TDM 0
#endif

__device__ __forceinline__ unsigned short f2bf(float f) {
  unsigned int u = __float_as_uint(f);
  unsigned int r = u + 0x7FFFu + ((u >> 16) & 1u);   // round-to-nearest-even
  return (unsigned short)(r >> 16);
}
__device__ __forceinline__ float bf2f(unsigned short b) {
  return __uint_as_float(((unsigned int)b) << 16);
}
// branch-free tanh/sigmoid on v_exp_f32; stable at both tails.
__device__ __forceinline__ float tanh_fast(float x) {
  float e = __expf(2.f * x);
  return 1.f - 2.f / (e + 1.f);
}
__device__ __forceinline__ float sigmoid_fast(float x) {
  return 1.f / (1.f + __expf(-x));
}

#if HAVE_TDM
// TDM gather-mode descriptor: pull 8 scattered rows (32-bit indices) of
// tile_dim0=100 f32 elements from a [NN x 100] tensor into LDS (rows packed
// back-to-back, 400B each).  Tracked by TENSORcnt.
__device__ __forceinline__ void tdm_gather8(unsigned lds_addr, const float* gbase,
                                            int i0, int i1, int i2, int i3,
                                            int i4, int i5, int i6, int i7) {
  unsigned long long ga = (unsigned long long)(uintptr_t)gbase;
  u32x4 g0 = {
    1u | (1u << 30) | (1u << 31),                     // count=1, 32b idx, gather
    lds_addr,                                         // D#.lds_addr
    (unsigned)(ga & 0xFFFFFFFFu),                     // global_addr[31:0]
    (unsigned)((ga >> 32) & 0x01FFFFFFu) | (2u << 30) // addr[56:32] | type=2
  };
  i32x8 g1 = {
    (int)(2u << 16),                                  // data_size=4B
    (int)(((unsigned)DD & 0xFFFFu) << 16),            // tensor_dim0 lo16
    (int)(((unsigned)NN & 0xFFFFu) << 16),            // t_dim0 hi | tensor_dim1 lo
    (int)(((unsigned)NN >> 16) | ((unsigned)DD << 16)), // tensor_dim1 hi | tile_dim0
    8,                                                // tile_dim1 = #indices
    DD,                                               // tensor_dim0_stride lo32
    0, 0
  };
  i32x4 g2 = { i0, i1, i2, i3 };
  i32x4 g3 = { i4, i5, i6, i7 };
#if defined(__clang_major__) && __clang_major__ >= 23
  i32x8 gz = { 0, 0, 0, 0, 0, 0, 0, 0 };
  __builtin_amdgcn_tensor_load_to_lds(g0, g1, g2, g3, gz, 0);
#else
  __builtin_amdgcn_tensor_load_to_lds(g0, g1, g2, g3, 0);
#endif
}
#endif

// ---------------------------------------------------------------------------
// Prep 1: convert + transpose the two 100x100 GEMM weights per head into
// bf16 [112][128] (N-major, K contiguous) so each lane's WMMA B-fragment is
// one contiguous 32B load.
// ---------------------------------------------------------------------------
__global__ void prep_weights(const float* __restrict__ aw1, const float* __restrict__ aw2,
                             const float* __restrict__ cw1, const float* __restrict__ cw2,
                             unsigned short* __restrict__ w1nT_a, unsigned short* __restrict__ w2T_a,
                             unsigned short* __restrict__ w1nT_c, unsigned short* __restrict__ w2T_c) {
  int idx = blockIdx.x * blockDim.x + threadIdx.x;
  if (idx >= 4 * 112 * 128) return;
  int mat = idx / (112 * 128);
  int rem = idx % (112 * 128);
  int n = rem / 128, k = rem % 128;
  float v = 0.f;
  if (n < 100 && k < 100) {
    switch (mat) {
      case 0: v = aw1[(100 + k) * DD + n]; break;   // neigh slice of actor_w1
      case 1: v = aw2[k * DD + n]; break;
      case 2: v = cw1[(100 + k) * DD + n]; break;
      default: v = cw2[k * DD + n]; break;
    }
  }
  unsigned short bv = f2bf(v);
  switch (mat) {
    case 0: w1nT_a[rem] = bv; break;
    case 1: w2T_a[rem]  = bv; break;
    case 2: w1nT_c[rem] = bv; break;
    default: w2T_c[rem] = bv; break;
  }
}

// ---------------------------------------------------------------------------
// Prep 2: rel_proj[r][n] = relation_embedding[r] @ w1[200:300, n]
// Only 1200 distinct relations -> hoisted out of the 256000-row GEMM.
// ---------------------------------------------------------------------------
__global__ void prep_rel(const float* __restrict__ rel_emb,
                         const float* __restrict__ aw1, const float* __restrict__ cw1,
                         float* __restrict__ rp_a, float* __restrict__ rp_c) {
  int idx = blockIdx.x * blockDim.x + threadIdx.x;
  if (idx >= 2 * RR * 112) return;
  int head = idx / (RR * 112);
  int rem  = idx % (RR * 112);
  int r = rem / 112, n = rem % 112;
  const float* w1 = head ? cw1 : aw1;
  float acc = 0.f;
  if (n < 100) {
    for (int i = 0; i < 100; i++) acc += rel_emb[r * DD + i] * w1[(200 + i) * DD + n];
  }
  (head ? rp_c : rp_a)[rem] = acc;
}

// ---------------------------------------------------------------------------
// Prep 3: state_proj[b][n] = b1[n] + graph_embedding[b] @ w1[0:100, n]
// ---------------------------------------------------------------------------
__global__ void prep_state(const float* __restrict__ graph,
                           const float* __restrict__ aw1, const float* __restrict__ cw1,
                           const float* __restrict__ ab1, const float* __restrict__ cb1,
                           float* __restrict__ sp_a, float* __restrict__ sp_c) {
  int idx = blockIdx.x * blockDim.x + threadIdx.x;
  if (idx >= 2 * BB * 112) return;
  int head = idx / (BB * 112);
  int rem  = idx % (BB * 112);
  int b = rem / 112, n = rem % 112;
  const float* w1 = head ? cw1 : aw1;
  const float* b1 = head ? cb1 : ab1;
  float acc = 0.f;
  if (n < 100) {
    acc = b1[n];
    for (int i = 0; i < 100; i++) acc += graph[b * DD + i] * w1[i * DD + n];
  }
  (head ? sp_c : sp_a)[rem] = acc;
}

// ---------------------------------------------------------------------------
// Main fused kernel: one block = one (batch, 16-row) tile.
// 128 threads = 4 wave32s.  TDM gather-mode DMA pulls the 16 scattered node
// rows into LDS (f32) while the waves set up indices/padding; then fused
// new_graph write + dot logit + bf16 A-tile, two WMMA GEMM layers per head,
// scalar layer 3.
// ---------------------------------------------------------------------------
__global__ __launch_bounds__(128)
void fused_main(const float* __restrict__ node_emb,
                const float* __restrict__ graph_emb,
                const float* __restrict__ origin_emb,
                const long long* __restrict__ entity_adj,
                const long long* __restrict__ relation_adj,
                const long long* __restrict__ nodes,
                const unsigned short* __restrict__ w1nT_a,
                const unsigned short* __restrict__ w2T_a,
                const unsigned short* __restrict__ w1nT_c,
                const unsigned short* __restrict__ w2T_c,
                const float* __restrict__ relproj_a, const float* __restrict__ relproj_c,
                const float* __restrict__ stateproj_a, const float* __restrict__ stateproj_c,
                const float* __restrict__ b2_a, const float* __restrict__ w3_a, const float* __restrict__ b3_a,
                const float* __restrict__ b2_c, const float* __restrict__ w3_c, const float* __restrict__ b3_c,
                float* __restrict__ out) {
  __shared__ __align__(16) float          a_f32[16][DD];   // TDM gather target
  __shared__ __align__(32) unsigned short a_sh[16][128];   // bf16 A tile (zero pad K 100..127)
  __shared__ __align__(32) unsigned short h1_sh[16][128];
  __shared__ __align__(32) unsigned short h2_sh[16][128];
  __shared__ int   s_nid[16];
  __shared__ int   s_rid[16];
  __shared__ int   s_rloc[16];
  __shared__ float s_dot[16];

  const int b     = blockIdx.y;
  const int rbase = blockIdx.x * 16;
  const int t     = threadIdx.x;
  const int lane  = t & 31;
  const int wave  = t >> 5;

  if (t == 0) {   // warm L2->WGP path for the shared weights
    __builtin_prefetch(w1nT_a, 0, 0);
    __builtin_prefetch(w2T_a, 0, 0);
    __builtin_prefetch(w1nT_c, 0, 0);
    __builtin_prefetch(w2T_c, 0, 0);
  }

#if HAVE_TDM
  // ---- issue async gather of the 16 scattered node rows (wave 0 only) ----
  if (wave == 0) {
    int nid_s[16];
#pragma unroll
    for (int i = 0; i < 16; i++) {
      int rl = rbase + i;
      int rc = rl < ROWS_PER_B ? rl : (ROWS_PER_B - 1);
      int m = rc / KK, k = rc % KK;
      long long mid = nodes[(long long)b * MM + m];
      nid_s[i] = __builtin_amdgcn_readfirstlane((int)entity_adj[mid * KK + k]);
    }
    unsigned lds0 = (unsigned)(uintptr_t)(void*)&a_f32[0][0];
    tdm_gather8(lds0, node_emb,
                nid_s[0], nid_s[1], nid_s[2], nid_s[3],
                nid_s[4], nid_s[5], nid_s[6], nid_s[7]);
    tdm_gather8(lds0 + 8u * DD * 4u, node_emb,
                nid_s[8], nid_s[9], nid_s[10], nid_s[11],
                nid_s[12], nid_s[13], nid_s[14], nid_s[15]);
  }
#endif

  // ---- phase A: indices, zero pad regions -------------------------------
  if (t < 16) {
    int rl = rbase + t;
    int rc = rl < ROWS_PER_B ? rl : (ROWS_PER_B - 1);   // clamp tail rows
    int m = rc / KK, k = rc % KK;
    long long mid = nodes[(long long)b * MM + m];
    s_nid[t]  = (int)entity_adj[mid * KK + k];
    s_rid[t]  = (int)relation_adj[mid * KK + k];
    s_rloc[t] = rl;
    s_dot[t]  = 0.f;
  }
  for (int i = t; i < 16 * 28; i += 128) {              // zero h1/h2 pad cols
    int r = i / 28, c = 100 + i % 28;
    h1_sh[r][c] = 0; h2_sh[r][c] = 0;
  }
#if HAVE_TDM
  __builtin_amdgcn_s_wait_tensorcnt(0);                 // wave0's DMA done
#endif
  __syncthreads();

  // ---- phase B: A tile from LDS (TDM) or global; fuse new_graph + dot ---
  {
    int rt = t >> 3;
    int c0 = (t & 7) * 16;
    int rl  = s_rloc[rt];
    bool valid = rl < ROWS_PER_B;
#if HAVE_TDM
    const float* src = (const float*)&a_f32[rt][0];
#else
    const float* src = node_emb + (long long)s_nid[rt] * DD;
#endif
    const float* geb = graph_emb + b * DD;
    const float* orb = origin_emb + b * DD;
    float* ng = out + OUT_NG + ((long long)b * ROWS_PER_B + rl) * DD;
    if (c0 < 96) {
      float part = 0.f;
#pragma unroll
      for (int q = 0; q < 4; q++) {
        float4 v = *(const float4*)(src + c0 + q * 4);
        float4 g = *(const float4*)(geb + c0 + q * 4);
        float4 o = *(const float4*)(orb + c0 + q * 4);
        float4 s;
        s.x = v.x + g.x; s.y = v.y + g.y; s.z = v.z + g.z; s.w = v.w + g.w;
        if (valid) *(float4*)(ng + c0 + q * 4) = s;
        part += o.x * s.x + o.y * s.y + o.z * s.z + o.w * s.w;
        a_sh[rt][c0 + q * 4 + 0] = f2bf(v.x);
        a_sh[rt][c0 + q * 4 + 1] = f2bf(v.y);
        a_sh[rt][c0 + q * 4 + 2] = f2bf(v.z);
        a_sh[rt][c0 + q * 4 + 3] = f2bf(v.w);
      }
      atomicAdd(&s_dot[rt], part);                      // ds_add_f32
    } else if (c0 == 96) {
      float4 v = *(const float4*)(src + 96);
      float4 g = *(const float4*)(geb + 96);
      float4 o = *(const float4*)(orb + 96);
      float4 s;
      s.x = v.x + g.x; s.y = v.y + g.y; s.z = v.z + g.z; s.w = v.w + g.w;
      if (valid) *(float4*)(ng + 96) = s;
      float part = o.x * s.x + o.y * s.y + o.z * s.z + o.w * s.w;
      a_sh[rt][96] = f2bf(v.x); a_sh[rt][97] = f2bf(v.y);
      a_sh[rt][98] = f2bf(v.z); a_sh[rt][99] = f2bf(v.w);
#pragma unroll
      for (int c = 100; c < 112; c++) a_sh[rt][c] = 0;
      atomicAdd(&s_dot[rt], part);
    } else {
#pragma unroll
      for (int c = 112; c < 128; c++) a_sh[rt][c] = 0;
    }
  }
  __syncthreads();

  // ---- phase C: dot logits (softmaxed later) ----------------------------
  if (t < 16) {
    int rl = s_rloc[t];
    if (rl < ROWS_PER_B) out[OUT_DOT + b * ROWS_PER_B + rl] = s_dot[t];
  }

  // WMMA fragment lane mapping (ISA 7.12.2): A lane=M (K split by half),
  // B lane=N (K contiguous from transposed weights), C/D VGPR j -> M=j+8*half.
  const int half = lane >> 4;
  const int nl   = lane & 15;

  for (int h = 0; h < 2; h++) {
    const unsigned short* w1h = h ? w1nT_c : w1nT_a;
    const unsigned short* w2h = h ? w2T_c  : w2T_a;
    const float* rph = h ? relproj_c : relproj_a;
    const float* sph = h ? stateproj_c : stateproj_a;
    const float* b2h = h ? b2_c : b2_a;
    const float* w3h = h ? w3_c : w3_a;
    const float  b3h = h ? b3_c[0] : b3_a[0];

    // ---- layer 1: neigh @ w1_neigh, epilogue adds state_proj + rel_proj --
    for (int nt = wave; nt < 7; nt += 4) {
      v8f acc = {};
      const unsigned short* bcol = w1h + (nt * 16 + nl) * 128 + half * 16;
#pragma unroll
      for (int kk = 0; kk < 4; kk++) {
        v8bf alo = *(const v8bf*)(&a_sh[nl][kk * 32 + half * 16]);
        v8bf ahi = *(const v8bf*)(&a_sh[nl][kk * 32 + half * 16 + 8]);
        v16bf af;
#pragma unroll
        for (int i = 0; i < 8; i++) { af[i] = alo[i]; af[i + 8] = ahi[i]; }
        v16bf bfrag = *(const v16bf*)(bcol + kk * 32);
        acc = __builtin_amdgcn_wmma_f32_16x16x32_bf16(false, af, false, bfrag,
                                                      (short)0, acc, false, false);
      }
#pragma unroll
      for (int j = 0; j < 8; j++) {
        int mrow = j + half * 8;
        int col  = nt * 16 + nl;
        if (col < 100) {
          float v = acc[j] + sph[b * 112 + col] + rph[s_rid[mrow] * 112 + col];
          h1_sh[mrow][col] = f2bf(tanh_fast(v));
        }
      }
    }
    __syncthreads();

    // ---- layer 2: h1 @ w2 -------------------------------------------------
    for (int nt = wave; nt < 7; nt += 4) {
      v8f acc = {};
      const unsigned short* bcol = w2h + (nt * 16 + nl) * 128 + half * 16;
#pragma unroll
      for (int kk = 0; kk < 4; kk++) {
        v8bf alo = *(const v8bf*)(&h1_sh[nl][kk * 32 + half * 16]);
        v8bf ahi = *(const v8bf*)(&h1_sh[nl][kk * 32 + half * 16 + 8]);
        v16bf af;
#pragma unroll
        for (int i = 0; i < 8; i++) { af[i] = alo[i]; af[i + 8] = ahi[i]; }
        v16bf bfrag = *(const v16bf*)(bcol + kk * 32);
        acc = __builtin_amdgcn_wmma_f32_16x16x32_bf16(false, af, false, bfrag,
                                                      (short)0, acc, false, false);
      }
#pragma unroll
      for (int j = 0; j < 8; j++) {
        int mrow = j + half * 8;
        int col  = nt * 16 + nl;
        if (col < 100) {
          float v = acc[j] + b2h[col];
          h2_sh[mrow][col] = f2bf(tanh_fast(v));
        }
      }
    }
    __syncthreads();

    // ---- layer 3: 100-dot + sigmoid --------------------------------------
    if (t < 16) {
      float acc = b3h;
      for (int c = 0; c < 100; c++) acc += bf2f(h2_sh[t][c]) * w3h[c];
      float sg = sigmoid_fast(acc);
      int rl = s_rloc[t];
      if (rl < ROWS_PER_B) out[(h ? OUT_Q : OUT_ACT) + b * ROWS_PER_B + rl] = sg;
    }
    __syncthreads();
  }
}

// ---------------------------------------------------------------------------
// In-place softmax over the 1000 logits of each batch row.
// ---------------------------------------------------------------------------
__global__ __launch_bounds__(256)
void softmax1000(float* __restrict__ logits) {
  __shared__ float red[256];
  int b = blockIdx.x;
  float* row = logits + b * ROWS_PER_B;
  int t = threadIdx.x;
  float mx = -3.4e38f;
  for (int i = t; i < ROWS_PER_B; i += 256) mx = fmaxf(mx, row[i]);
  red[t] = mx; __syncthreads();
  for (int s = 128; s > 0; s >>= 1) { if (t < s) red[t] = fmaxf(red[t], red[t + s]); __syncthreads(); }
  mx = red[0]; __syncthreads();
  float sum = 0.f;
  for (int i = t; i < ROWS_PER_B; i += 256) { float e = __expf(row[i] - mx); row[i] = e; sum += e; }
  red[t] = sum; __syncthreads();
  for (int s = 128; s > 0; s >>= 1) { if (t < s) red[t] += red[t + s]; __syncthreads(); }
  float inv = 1.f / red[0];
  for (int i = t; i < ROWS_PER_B; i += 256) row[i] *= inv;
}

// ---------------------------------------------------------------------------
extern "C" void kernel_launch(void* const* d_in, const int* in_sizes, int n_in,
                              void* d_out, int out_size, void* d_ws, size_t ws_size,
                              hipStream_t stream) {
  (void)in_sizes; (void)n_in; (void)out_size; (void)ws_size;
  const float* node_emb = (const float*)d_in[0];
  const float* rel_emb  = (const float*)d_in[1];
  const float* origin   = (const float*)d_in[2];
  const float* graph    = (const float*)d_in[3];
  const float* aw1 = (const float*)d_in[4];
  const float* ab1 = (const float*)d_in[5];
  const float* aw2 = (const float*)d_in[6];
  const float* ab2 = (const float*)d_in[7];
  const float* aw3 = (const float*)d_in[8];
  const float* ab3 = (const float*)d_in[9];
  const float* cw1 = (const float*)d_in[10];
  const float* cb1 = (const float*)d_in[11];
  const float* cw2 = (const float*)d_in[12];
  const float* cb2 = (const float*)d_in[13];
  const float* cw3 = (const float*)d_in[14];
  const float* cb3 = (const float*)d_in[15];
  const long long* eadj  = (const long long*)d_in[16];
  const long long* radj  = (const long long*)d_in[17];
  const long long* nodes = (const long long*)d_in[18];
  float* out = (float*)d_out;

  // workspace layout (bf16 weight pair per head kept adjacent)
  unsigned short* w1nT_a = (unsigned short*)d_ws;       // [112*128] bf16
  unsigned short* w2T_a  = w1nT_a + 112 * 128;
  unsigned short* w1nT_c = w2T_a  + 112 * 128;
  unsigned short* w2T_c  = w1nT_c + 112 * 128;
  float* relproj_a   = (float*)(w2T_c + 112 * 128);     // [1200*112] f32
  float* relproj_c   = relproj_a + RR * 112;
  float* stateproj_a = relproj_c + RR * 112;            // [256*112] f32
  float* stateproj_c = stateproj_a + BB * 112;

  prep_weights<<<(4 * 112 * 128 + 255) / 256, 256, 0, stream>>>(
      aw1, aw2, cw1, cw2, w1nT_a, w2T_a, w1nT_c, w2T_c);
  prep_rel<<<(2 * RR * 112 + 255) / 256, 256, 0, stream>>>(
      rel_emb, aw1, cw1, relproj_a, relproj_c);
  prep_state<<<(2 * BB * 112 + 255) / 256, 256, 0, stream>>>(
      graph, aw1, cw1, ab1, cb1, stateproj_a, stateproj_c);
  fused_main<<<dim3(TILES_PER_B, BB), 128, 0, stream>>>(
      node_emb, graph, origin, eadj, radj, nodes,
      w1nT_a, w2T_a, w1nT_c, w2T_c,
      relproj_a, relproj_c, stateproj_a, stateproj_c,
      ab2, aw3, ab3, cb2, cw3, cb3, out);
  softmax1000<<<BB, 256, 0, stream>>>(out + OUT_DOT);
}
